// Capsule_14688788152928
// MI455X (gfx1250) — matrix operand order repaired and checked by reading, not compiled, not run on hardware
//
#include <hip/hip_runtime.h>
#include <hip/hip_bf16.h>

typedef __attribute__((ext_vector_type(16))) __bf16 v16bf;
typedef __attribute__((ext_vector_type(8)))  float  v8f;
typedef __attribute__((ext_vector_type(4)))  float  v4f;

#define B_ 64
#define N_ 2048
#define I_ 16
#define J_ 32
#define K_ 32

union Frag16 { unsigned u[8]; v16bf v; };

// bf16 A-matrix 16x32 lane layout (ISA 7.12.2): lane L -> m = L%16, half = L/16
// VGPR v: K = 2v + 16*half (contiguous order chosen here, not the ISA table order; matches B)
__device__ __forceinline__ int a_kk(int v, int h) {
  return 16 * h + 2 * v;
}
// bf16 B-matrix 32x16 lane layout: lane L -> col = L%16 ; VGPR v: K = 2v + 16*half
__device__ __forceinline__ int b_kk(int v, int h) {
  return 16 * h + 2 * v;
}

__device__ __forceinline__ unsigned lds_off(const void* p) {
  // flat LDS pointer: addr[31:0] is the allocation-relative LDS byte offset
  return (unsigned)(unsigned long long)p;
}

__global__ __launch_bounds__(256) void kzero(float* p, int n) {
  int i = blockIdx.x * 256 + threadIdx.x;
  if (i < n) p[i] = 0.0f;
}

// one-time W fp32 -> bf16; NT-load the fp32 source so only bf16 W lands in L2
__global__ __launch_bounds__(256) void kcvt(const float* __restrict__ W,
                                            __bf16* __restrict__ Wb) {
  const int i = (blockIdx.x * 256 + threadIdx.x) * 4;
  const v4f f = __builtin_nontemporal_load((const v4f*)(W + i));
  Wb[i + 0] = (__bf16)f[0];
  Wb[i + 1] = (__bf16)f[1];
  Wb[i + 2] = (__bf16)f[2];
  Wb[i + 3] = (__bf16)f[3];
}

// s[b,j,k] += sum_{n in chunk, i} cval * x[b,n,i] * W[j,n,k,i]
// grid = (J, N/128), block = 256 (8 waves: wave w -> m-tile w&3, k-tile w>>2)
// W tile ring: 4-deep, async-filled 2 steps ahead (bf16 path); A tiles 2-deep.
__global__ __launch_bounds__(256) void ksum(const float* __restrict__ x,
                                            const float* __restrict__ W,
                                            const __bf16* __restrict__ Wb,
                                            const float* __restrict__ c,  // [J][N][B]
                                            float* __restrict__ s,        // [B][J][K]
                                            int uniform) {
  __shared__ __bf16 Al[2][64][36];   // A: [b][kk], kk = dn*16+i (padded rows)
  __shared__ __bf16 Wl[4][1024];     // raw W tile: [dn][kc][i], 2KB each
  __shared__ __bf16 Wdummy[1024];

  const int j   = blockIdx.x;
  const int n0  = blockIdx.y * 128;
  const int tid = threadIdx.x;
  const int lane = tid & 31, w = tid >> 5;
  const int mt = w & 3, kt = w >> 2;
  const int half = lane >> 4, lm = lane & 15;
  const bool use_bf = (Wb != nullptr);

  const size_t wrow = ((size_t)j * N_ + n0) * (K_ * I_);  // elems, step = 2*K*I

  if (use_bf && w < 4) {  // prologue: async-issue W(0), W(1)
#pragma unroll
    for (int t = 0; t < 2; ++t) {
      const unsigned l = lds_off(&Wl[t][w * 256 + lane * 8]);
      const unsigned long long g =
          (unsigned long long)(Wb + wrow + (size_t)t * 1024 + w * 256 + lane * 8);
      asm volatile("global_load_async_to_lds_b128 %0, %1, off"
                   :: "v"(l), "v"(g) : "memory");
    }
  }

  v8f acc = {0.f, 0.f, 0.f, 0.f, 0.f, 0.f, 0.f, 0.f};

  for (int t = 0; t < 64; ++t) {
    const int n = n0 + 2 * t;
    {  // stage A = c * x into Al[t&1]: b = tid/4, dn = (tid%4)/2, i0 = (tid%2)*8
      const int b = tid >> 2, g = tid & 3, dn = g >> 1, i0 = (g & 1) * 8;
      const float cv = uniform ? (1.0f / 32.0f)
                               : c[((size_t)j * N_ + n + dn) * B_ + b];
      const float* xp = x + ((size_t)b * N_ + n + dn) * I_ + i0;
#pragma unroll
      for (int q = 0; q < 8; ++q)
        Al[t & 1][b][dn * 16 + i0 + q] = (__bf16)(cv * xp[q]);
    }
    if (use_bf) {
      if (w < 4) {  // async-issue W(t+2) into ring slot (t+2)&3 (dummy at tail)
        const bool real = (t + 2) < 64;
        __bf16* dst = real ? &Wl[(t + 2) & 3][0] : &Wdummy[0];
        const __bf16* src = Wb + wrow + (size_t)(real ? t + 2 : 63) * 1024;
        const unsigned l = lds_off(dst + w * 256 + lane * 8);
        const unsigned long long g =
            (unsigned long long)(src + w * 256 + lane * 8);
        asm volatile("global_load_async_to_lds_b128 %0, %1, off"
                     :: "v"(l), "v"(g) : "memory");
        // 3 outstanding max; <=2 left  =>  W(t) has landed (in-order completion)
        asm volatile("s_wait_asynccnt 0x2" ::: "memory");
      }
    } else if (tid < 128) {  // fp32 fallback: sync cvt into Wl[t&3], raw layout
      const int e0 = tid * 8;
      const float* wp = W + wrow + (size_t)t * 1024 + e0;
#pragma unroll
      for (int q = 0; q < 8; ++q) Wl[t & 3][e0 + q] = (__bf16)wp[q];
      if (t + 1 < 64) __builtin_prefetch(wp + 1024, 0, 1);
    }
    __syncthreads();

    Frag16 fa, fb;
#pragma unroll
    for (int v = 0; v < 8; ++v) {
      fa.u[v] = *(const unsigned*)&Al[t & 1][mt * 16 + lm][a_kk(v, half)];
      // raw layout: kk = 16*half + 2v -> dn = half, i = 2v (pair contiguous)
      fb.u[v] = *(const unsigned*)&Wl[t & 3][half * 512 + (kt * 16 + lm) * 16 + 2 * v];
    }
    acc = __builtin_amdgcn_wmma_f32_16x16x32_bf16(false, fa.v, false, fb.v,
                                                  (short)0, acc, false, false);
    // no trailing barrier: A is 2-deep, W ring is 4-deep (write distance 3)
  }
  // D tile: lane L, vgpr r -> M = r + 8*half, N = L%16
#pragma unroll
  for (int r = 0; r < 8; ++r) {
    const int b = mt * 16 + r + 8 * half;
    const int k = kt * 16 + lm;
    atomicAdd(&s[((size_t)b * J_ + j) * K_ + k], acc[r]);
  }
}

// squash: v = s * ||s||^2/(1+||s||^2)/sqrt(||s||^2+eps); zero s for next pass
__global__ __launch_bounds__(256) void kv(float* __restrict__ s,
                                          __bf16* __restrict__ vbf,
                                          float* __restrict__ out, int final_) {
  const int p = blockIdx.x * 8 + (threadIdx.x >> 5);  // p = b*J + j
  const int lane = threadIdx.x & 31;                  // lane = k (K==32)
  const size_t idx = (size_t)p * K_ + lane;
  const float sv = s[idx];
  float sq = sv * sv;
#pragma unroll
  for (int d = 1; d < 32; d <<= 1) sq += __shfl_xor(sq, d);
  const float scale = sq / (1.0f + sq) * rsqrtf(sq + 1e-7f);
  const float o = scale * sv;
  vbf[idx] = (__bf16)o;
  if (final_) out[idx] = o;
  s[idx] = 0.0f;
}

// b[b,j,n] += sum_k v[b,j,k]*u_hat[b,j,n,k]; then c = softmax_j(b)
// u_hat recomputed: T[b,i] = sum_k v[b,k]*W[j,n,k,i]; delta = sum_i T[b,i]*x[b,n,i]
__global__ __launch_bounds__(256) void krouting(const float* __restrict__ x,
                                                const float* __restrict__ W,
                                                const __bf16* __restrict__ Wb,
                                                const __bf16* __restrict__ vbf,
                                                float* __restrict__ bstate,  // [J][N][B]
                                                float* __restrict__ cglob) { // [J][N][B]
  extern __shared__ char smem[];
  float*  xf = (float*)smem;                 // [8][64][16]
  float*  bn = xf + 8 * 64 * 16;             // [8][64][32]
  __bf16* va = (__bf16*)(bn + 8 * 64 * 32);  // [64][36]
  __bf16* wt = va + 64 * 36;                 // [2][16][36]  W^T tiles [dnn][i][k]

  const int n0  = blockIdx.x * 8;
  const int tid = threadIdx.x;
  const int lane = tid & 31, w = tid >> 5;
  const int mtl = w & 3, dnn_w = w >> 2;
  const int half = lane >> 4, lm = lane & 15;

  {  // stage x chunk once
    const int nn = tid >> 5, b0 = (tid & 31) * 2;
#pragma unroll
    for (int db = 0; db < 2; ++db) {
      const float* xp = x + ((size_t)(b0 + db) * N_ + n0 + nn) * I_;
#pragma unroll
      for (int i = 0; i < 16; ++i) xf[(nn * 64 + b0 + db) * 16 + i] = xp[i];
    }
  }

  for (int j = 0; j < J_; ++j) {
    __syncthreads();
    if (tid < 128) {  // stage v[.,j,.]: b = t/2, k0 = (t%2)*16
      const int b = tid >> 1, k0 = (tid & 1) * 16;
      const unsigned* g = (const unsigned*)(vbf + (((size_t)b * J_ + j) * K_ + k0));
#pragma unroll
      for (int q = 0; q < 8; ++q) *(unsigned*)&va[b * 36 + k0 + 2 * q] = g[q];
    }
    for (int nn0 = 0; nn0 < 8; nn0 += 2) {
      __syncthreads();
      {  // stage wt[dnn][i][k] = W[j, n0+nn0+dnn, k, i] (transpose to k-contig)
        const int lin = tid * 4;
        const int dnn = lin >> 9, rem = lin & 511;
        const int k = rem >> 4, i0 = rem & 15;
        const size_t base = (((size_t)j * N_ + n0 + nn0 + dnn) * K_ + k) * I_ + i0;
        if (Wb) {
          const __bf16* wp = Wb + base;
#pragma unroll
          for (int q = 0; q < 4; ++q) wt[(dnn * 16 + i0 + q) * 36 + k] = wp[q];
        } else {
          const float* wp = W + base;
#pragma unroll
          for (int q = 0; q < 4; ++q) wt[(dnn * 16 + i0 + q) * 36 + k] = (__bf16)wp[q];
        }
      }
      __syncthreads();

      const int nn = nn0 + dnn_w;
      const int n = n0 + nn;
      Frag16 fa, fb;
#pragma unroll
      for (int v = 0; v < 8; ++v) {
        fa.u[v] = *(const unsigned*)&va[(mtl * 16 + lm) * 36 + a_kk(v, half)];
        fb.u[v] = *(const unsigned*)&wt[(dnn_w * 16 + lm) * 36 + b_kk(v, half)];
      }
      v8f T = {0.f, 0.f, 0.f, 0.f, 0.f, 0.f, 0.f, 0.f};
      T = __builtin_amdgcn_wmma_f32_16x16x32_bf16(false, fa.v, false, fb.v,
                                                  (short)0, T, false, false);
      float red[8];
#pragma unroll
      for (int r = 0; r < 8; ++r) {
        const int b = mtl * 16 + r + 8 * half;
        float pv = T[r] * xf[(nn * 64 + b) * 16 + lm];
#pragma unroll
        for (int d = 1; d < 16; d <<= 1) pv += __shfl_xor(pv, d);
        red[r] = pv;
      }
      if (lm == 0) {
#pragma unroll
        for (int r = 0; r < 8; ++r) {
          const int b = mtl * 16 + r + 8 * half;
          const size_t bi = ((size_t)j * N_ + n) * B_ + b;
          const float bv = bstate[bi] + red[r];
          bstate[bi] = bv;
          bn[(nn * 64 + b) * 32 + j] = bv;
        }
      }
    }
  }
  __syncthreads();
  for (int p = tid; p < 8 * 64; p += 256) {  // softmax over j per (nn, b)
    const int nn = p >> 6, b = p & 63;
    const float* row = &bn[(nn * 64 + b) * 32];
    float m = row[0];
#pragma unroll
    for (int j = 1; j < 32; ++j) m = fmaxf(m, row[j]);
    float ssum = 0.f;
#pragma unroll
    for (int j = 0; j < 32; ++j) ssum += __expf(row[j] - m);
    const float inv = 1.0f / ssum;
    const int n = n0 + nn;
#pragma unroll
    for (int j = 0; j < 32; ++j)
      cglob[((size_t)j * N_ + n) * B_ + b] = __expf(row[j] - m) * inv;
  }
}

extern "C" void kernel_launch(void* const* d_in, const int* in_sizes, int n_in,
                              void* d_out, int out_size, void* d_ws, size_t ws_size,
                              hipStream_t stream) {
  const float* x = (const float*)d_in[0];  // [B,N,I]
  const float* W = (const float*)d_in[1];  // [J,N,K,I]
  float* out = (float*)d_out;              // [B,J,K]

  char* ws = (char*)d_ws;
  const size_t JNB = (size_t)J_ * N_ * B_;        // 4M
  const size_t BJK = (size_t)B_ * J_ * K_;        // 64K
  float*  bstate = (float*)ws;                    // 16 MB
  float*  cbuf   = (float*)(ws + JNB * 4);        // 16 MB
  float*  sbuf   = (float*)(ws + JNB * 8);        // 256 KB
  __bf16* vbf    = (__bf16*)(ws + JNB * 8 + BJK * 4);
  const size_t base = JNB * 8 + BJK * 4 + BJK * 2;
  const size_t welts = (size_t)J_ * N_ * K_ * I_;           // 33.5M
  const bool use_bf = ws_size >= base + welts * 2;          // need ~101 MB
  __bf16* Wb = use_bf ? (__bf16*)(ws + base) : nullptr;

  kzero<<<(int)((JNB + 255) / 256), 256, 0, stream>>>(bstate, (int)JNB);
  kzero<<<(int)((BJK + 255) / 256), 256, 0, stream>>>(sbuf, (int)BJK);
  if (use_bf)
    kcvt<<<(int)(welts / 4 / 256), 256, 0, stream>>>(W, Wb);

  const dim3 gs(J_, N_ / 128);
  const size_t ka_smem =
      (size_t)(8 * 64 * 16 + 8 * 64 * 32) * sizeof(float) +
      (size_t)(64 * 36 + 2 * 16 * 36) * sizeof(__bf16);  // ~105 KB

  // round 0: uniform c
  ksum<<<gs, 256, 0, stream>>>(x, W, Wb, cbuf, sbuf, 1);
  kv<<<(B_ * J_) / 8, 256, 0, stream>>>(sbuf, vbf, out, 0);
  // rounds 1..2
  for (int it = 0; it < 2; ++it) {
    krouting<<<N_ / 8, 256, ka_smem, stream>>>(x, W, Wb, vbf, bstate, cbuf);
    ksum<<<gs, 256, 0, stream>>>(x, W, Wb, cbuf, sbuf, 0);
    kv<<<(B_ * J_) / 8, 256, 0, stream>>>(sbuf, vbf, out, it == 1);
  }
}
